// MosfetFNO_25202868093486
// MI455X (gfx1250) — matrix operand order repaired and checked by reading, not compiled, not run on hardware
//
#include <hip/hip_runtime.h>
#include <math.h>

typedef __bf16 bf16;
typedef __attribute__((ext_vector_type(16))) __bf16 v16bf;
typedef __attribute__((ext_vector_type(8)))  __bf16 v8bf;
typedef __attribute__((ext_vector_type(4)))  __bf16 v4bf;
typedef __attribute__((ext_vector_type(8)))  float   v8f;
typedef __attribute__((ext_vector_type(4)))  float   v4f;

#define B_     128
#define T_     4096
#define P_     64
#define E_     32
#define HEMB   128
#define CIN    36
#define HL     128
#define HP     128
#define NL     4
#define TP     4506     // T + round(0.1*T)
#define TPK    4512     // padded to multiple of 32 for WMMA K loops
#define NMODE2 128      // 64 modes * {re,im}

// fast silu: x * rcp(1+e^-x)  (v_exp + v_rcp, no IEEE div chain)
__device__ __forceinline__ float siluf(float x){
  return x * __builtin_amdgcn_rcpf(1.f + __expf(-x));
}
__device__ __forceinline__ float geluf(float x){
  float x3 = x*x*x;
  return 0.5f * x * (1.f + tanhf(0.7978845608028654f * (x + 0.044715f * x3)));
}

// ---- WMMA fragment helpers (bf16 16x16x32, wave32) -------------------------
// A (16x32, MxK) row-major: lane(&15)=row, half=lane>>4 selects K 0..7/16..23 vs 8..15/24..31
__device__ __forceinline__ v16bf frag_a(const bf16* base, int ld, int row0, int k0){
  int lane = threadIdx.x & 31;
  const bf16* p = base + (size_t)(row0 + (lane & 15)) * ld + k0 + (lane >> 4) * 8;
  v8bf lo = *(const v8bf*)p;
  v8bf hi = *(const v8bf*)(p + 16);
  v16bf r;
  #pragma unroll
  for (int i = 0; i < 8; ++i){ r[i] = lo[i]; r[i + 8] = hi[i]; }
  return r;
}
// B (32x16, KxN) row-major: lane(&15)=N column, half=lane>>4 selects K 0..7/16..23 vs 8..15/24..31
__device__ __forceinline__ v16bf frag_b(const bf16* base, int ld, int k0, int n0){
  int lane = threadIdx.x & 31;
  const bf16* p = base + (size_t)(k0 + (lane >> 4) * 8) * ld + n0 + (lane & 15);
  v16bf r;
  #pragma unroll
  for (int i = 0; i < 8; ++i){ r[i] = p[(size_t)i * ld]; r[i + 8] = p[(size_t)(i + 16) * ld]; }
  return r;
}
#define WMMA_BF16(a,b,c) \
  __builtin_amdgcn_wmma_f32_16x16x32_bf16(false,(a),false,(b),(short)0,(c),false,false)

// ---- tiny embedding MLP: (B,64)->gelu 128->gelu 128->tanh 32 ---------------
__global__ void k_embed(const float* __restrict__ pp,
                        const float* __restrict__ w1, const float* __restrict__ b1,
                        const float* __restrict__ w2, const float* __restrict__ b2,
                        const float* __restrict__ w3, const float* __restrict__ b3,
                        float* __restrict__ embout){
  __shared__ float sp[P_], h1[HEMB], h2[HEMB];
  int b = blockIdx.x, tid = threadIdx.x;
  if (tid < P_) sp[tid] = pp[b * P_ + tid];
  __syncthreads();
  float a = b1[tid];
  for (int p = 0; p < P_; ++p) a += sp[p] * w1[p * HEMB + tid];
  h1[tid] = geluf(a);
  __syncthreads();
  a = b2[tid];
  for (int p = 0; p < HEMB; ++p) a += h1[p] * w2[p * HEMB + tid];
  h2[tid] = geluf(a);
  __syncthreads();
  if (tid < E_){
    a = b3[tid];
    for (int p = 0; p < HEMB; ++p) a += h2[p] * w3[p * E_ + tid];
    embout[b * E_ + tid] = tanhf(a);
  }
}

// ---- convert small weight matrices to bf16 A-operands ----------------------
__global__ void k_wconv(const float* __restrict__ lw1, const float* __restrict__ lw2,
                        const float* __restrict__ skw, const float* __restrict__ pw1,
                        bf16* w1t, bf16* w2t, bf16* skbf, bf16* p1t){
  for (int i = threadIdx.x; i < HL * 64; i += blockDim.x){     // W1^T 128x64 (K padded 36->64)
    int d = i >> 6, c = i & 63;
    w1t[i] = (c < CIN) ? (bf16)lw1[c * HL + d] : (bf16)0.f;
  }
  for (int i = threadIdx.x; i < 64 * HL; i += blockDim.x){     // W2^T 64x128
    int r = i >> 7, c = i & 127;
    w2t[i] = (bf16)lw2[c * 64 + r];
  }
  for (int i = threadIdx.x; i < NL * 64 * 64; i += blockDim.x) // skip_w[l][o][c] direct
    skbf[i] = (bf16)skw[i];
  for (int i = threadIdx.x; i < HP * 64; i += blockDim.x){     // proj_w1^T 128x64
    int h = i >> 6, c = i & 63;
    p1t[i] = (bf16)pw1[c * HP + h];
  }
}

// ---- DFT bases: fwd[t][2m]=cos, [2m+1]=-sin ; inv[2m][t]=c_m cos, [2m+1]=-c_m sin
__global__ void k_basis(bf16* __restrict__ fwd, bf16* __restrict__ inv){
  int idx = blockIdx.x * blockDim.x + threadIdx.x;
  if (idx >= TPK * NMODE2) return;
  int t = idx >> 7, n = idx & 127;
  int m = n >> 1; int im = n & 1;
  float c = 0.f, s = 0.f;
  if (t < TP){
    int r = (int)(((long long)m * t) % TP);              // exact phase reduction
    float ph = 6.283185307179586f * (float)r / (float)TP;
    __sincosf(ph, &s, &c);
  }
  float cm = ((m == 0) ? 1.f : 2.f) / (float)TP;
  fwd[t * NMODE2 + n] = (bf16)(im ? -s : c);
  inv[n * TPK + t]    = (bf16)(im ? -cm * s : cm * c);
}

// ---- build concatenated lift input (B,64,T) bf16: v_terminals + broadcast emb
__global__ void k_xcat(const float* __restrict__ vt, const float* __restrict__ emb,
                       bf16* __restrict__ xcat){
  int c = blockIdx.x, b = blockIdx.y;
  bf16* row = xcat + ((size_t)b * 64 + c) * T_;
  if (c < 4){
    const float* src = vt + ((size_t)b * 4 + c) * T_;
    for (int t = threadIdx.x; t < T_; t += blockDim.x) row[t] = (bf16)src[t];
  } else if (c < CIN){
    bf16 v = (bf16)emb[b * E_ + (c - 4)];
    for (int t = threadIdx.x; t < T_; t += blockDim.x) row[t] = v;
  } else {
    for (int t = threadIdx.x; t < T_; t += blockDim.x) row[t] = (bf16)0.f;
  }
}

// ---- lift: x = W2^T silu(W1^T xcat + b1) + b2 ; fused via LDS tile ---------
__global__ void k_lift(const bf16* __restrict__ xcat, const bf16* __restrict__ w1t,
                       const bf16* __restrict__ w2t, const float* __restrict__ b1,
                       const float* __restrict__ b2, float* __restrict__ xbuf){
  __shared__ __attribute__((aligned(16))) bf16 h1[HL * 16];
  int b = blockIdx.y, t0 = blockIdx.x * 16;
  int w = threadIdx.x >> 5, lane = threadIdx.x & 31;
  int col = lane & 15, rb = (lane >> 4) * 8;
  const bf16* xc = xcat + (size_t)b * 64 * T_;
  v8f acc = {};
  #pragma unroll
  for (int ki = 0; ki < 2; ++ki){
    v16bf a  = frag_a(w1t, 64, 16 * w, 32 * ki);
    v16bf bb = frag_b(xc, T_, 32 * ki, t0);
    acc = WMMA_BF16(a, bb, acc);
  }
  #pragma unroll
  for (int v = 0; v < 8; ++v){
    int row = 16 * w + rb + v;
    h1[row * 16 + col] = (bf16)siluf(acc[v] + b1[row]);
  }
  __syncthreads();
  if (w < 4){
    v8f acc2 = {};
    #pragma unroll
    for (int ki = 0; ki < 4; ++ki){
      v16bf a  = frag_a(w2t, HL, 16 * w, 32 * ki);
      v16bf bb = frag_b(h1, 16, 32 * ki, 0);
      acc2 = WMMA_BF16(a, bb, acc2);
    }
    #pragma unroll
    for (int v = 0; v < 8; ++v){
      int row = 16 * w + rb + v;
      xbuf[((size_t)b * 64 + row) * TPK + t0 + col] = acc2[v] + b2[row];
    }
  }
}

__global__ void k_zeropad(float* __restrict__ xbuf){
  int idx = blockIdx.x * blockDim.x + threadIdx.x;
  int row = idx / (TPK - T_);
  int t   = T_ + idx % (TPK - T_);
  if (row < B_ * 64) xbuf[(size_t)row * TPK + t] = 0.f;
}

// vectorized: float4 in, 4x bf16 out
__global__ void k_silu(const float* __restrict__ xbuf, bf16* __restrict__ hb){
  size_t i4 = ((size_t)blockIdx.x * blockDim.x + threadIdx.x) * 4;
  v4f x = *(const v4f*)(xbuf + i4);
  v4bf o;
  #pragma unroll
  for (int j = 0; j < 4; ++j) o[j] = (bf16)siluf(x[j]);
  *(v4bf*)(hb + i4) = o;
}

// ---- forward DFT: Xspec[b][c][:] = silu(x)[b][c][:] @ fwd ------------------
// one block per batch, 4 waves = 4 m-tiles; each wave holds all 8 n-tile
// accumulators so every A fragment feeds 8 WMMAs (hb streamed exactly once).
__global__ void k_dftf(const bf16* __restrict__ hb, const bf16* __restrict__ fwd,
                       float* __restrict__ xspec){
  int b = blockIdx.x;
  int w = threadIdx.x >> 5, lane = threadIdx.x & 31;
  int col = lane & 15, rb = (lane >> 4) * 8;
  const bf16* hbb = hb + (size_t)b * 64 * TPK;
  v8f acc[8] = {};
  for (int ki = 0; ki < TPK / 32; ++ki){
    v16bf a = frag_a(hbb, TPK, 16 * w, 32 * ki);
    #pragma unroll
    for (int nt = 0; nt < 8; ++nt){
      v16bf bb = frag_b(fwd, NMODE2, 32 * ki, 16 * nt);
      acc[nt] = WMMA_BF16(a, bb, acc[nt]);
    }
  }
  #pragma unroll
  for (int nt = 0; nt < 8; ++nt){
    #pragma unroll
    for (int v = 0; v < 8; ++v){
      int row = 16 * w + rb + v;
      xspec[((size_t)b * 64 + row) * NMODE2 + 16 * nt + col] = acc[nt][v];
    }
  }
}

// ---- per-mode complex channel mix: Y[b,o,m] = sum_c X[b,c,m] * W[c,o,m] ----
__global__ void k_specmix(const float* __restrict__ xspec, const float* __restrict__ swr,
                          const float* __restrict__ swi, bf16* __restrict__ ybf, int l){
  __shared__ float Wr[64 * 64], Wi[64 * 64];
  int m = blockIdx.x;
  for (int i = threadIdx.x; i < 4096; i += blockDim.x){
    Wr[i] = swr[((size_t)l * 4096 + i) * 64 + m];
    Wi[i] = swi[((size_t)l * 4096 + i) * 64 + m];
  }
  __syncthreads();
  for (int pair = threadIdx.x; pair < B_ * 64; pair += blockDim.x){
    int b = pair >> 6, o = pair & 63;
    float yr = 0.f, yi = 0.f;
    const float* xb = xspec + (size_t)b * 64 * NMODE2 + 2 * m;
    for (int c = 0; c < 64; ++c){
      float xr = xb[c * NMODE2], xi = xb[c * NMODE2 + 1];
      float wr = Wr[c * 64 + o], wi = Wi[c * 64 + o];
      yr += xr * wr - xi * wi;
      yi += xr * wi + xi * wr;
    }
    ybf[((size_t)b * 64 + o) * NMODE2 + 2 * m]     = (bf16)yr;
    ybf[((size_t)b * 64 + o) * NMODE2 + 2 * m + 1] = (bf16)yi;
  }
}

// ---- fused skip-matmul + inverse DFT, in-place layer update ----------------
// 2 waves/block, each wave owns one t-tile and all 4 m-tiles: every streamed
// B fragment (hb tile / inv-basis tile) feeds 4 WMMAs.
__global__ void k_layerout(const bf16* __restrict__ hb, const bf16* __restrict__ skbf,
                           const float* __restrict__ skb, const bf16* __restrict__ ybf,
                           const bf16* __restrict__ inv, float* __restrict__ xbuf, int l){
  int b = blockIdx.y;
  int w = threadIdx.x >> 5, lane = threadIdx.x & 31;
  int col = lane & 15, rb = (lane >> 4) * 8;
  int t0 = (blockIdx.x * 2 + w) * 16;
  const bf16* hbb = hb + (size_t)b * 64 * TPK;
  const bf16* yb  = ybf + (size_t)b * 64 * NMODE2;
  const bf16* ska = skbf + l * 64 * 64;
  v8f acc[4] = {};
  #pragma unroll
  for (int ki = 0; ki < 2; ++ki){                   // skip: skip_w @ silu(x)
    v16bf bb = frag_b(hbb, TPK, 32 * ki, t0);
    #pragma unroll
    for (int mt = 0; mt < 4; ++mt){
      v16bf a = frag_a(ska, 64, 16 * mt, 32 * ki);
      acc[mt] = WMMA_BF16(a, bb, acc[mt]);
    }
  }
  #pragma unroll
  for (int ki = 0; ki < 4; ++ki){                   // inverse DFT: Y @ inv-basis
    v16bf bb = frag_b(inv, TPK, 32 * ki, t0);
    #pragma unroll
    for (int mt = 0; mt < 4; ++mt){
      v16bf a = frag_a(yb, NMODE2, 16 * mt, 32 * ki);
      acc[mt] = WMMA_BF16(a, bb, acc[mt]);
    }
  }
  int t = t0 + col;
  #pragma unroll
  for (int mt = 0; mt < 4; ++mt){
    #pragma unroll
    for (int v = 0; v < 8; ++v){
      int row = 16 * mt + rb + v;
      float val = acc[mt][v] + skb[l * 64 + row];
      xbuf[((size_t)b * 64 + row) * TPK + t] = (t < TP) ? val : 0.f;  // keep K-pad zero
    }
  }
}

// ---- projection head: out = W2 . silu(W1^T x + b1) + b2 (cropped to T) -----
__global__ void k_proj(const float* __restrict__ xbuf, const bf16* __restrict__ p1t,
                       const float* __restrict__ b1, const float* __restrict__ w2,
                       const float* __restrict__ b2, float* __restrict__ out){
  __shared__ __attribute__((aligned(16))) bf16 xt[64 * 16];
  __shared__ float hid[HP * 16];
  __shared__ float red[16 * 16];
  int b = blockIdx.y, t0 = blockIdx.x * 16;
  int tid = threadIdx.x;
  for (int i = tid; i < 64 * 16; i += blockDim.x){
    int c = i >> 4, t = i & 15;
    xt[i] = (bf16)xbuf[((size_t)b * 64 + c) * TPK + t0 + t];
  }
  __syncthreads();
  int w = tid >> 5, lane = tid & 31;
  int col = lane & 15, rb = (lane >> 4) * 8;
  v8f acc = {};
  #pragma unroll
  for (int ki = 0; ki < 2; ++ki){
    v16bf a  = frag_a(p1t, 64, 16 * w, 32 * ki);
    v16bf bb = frag_b(xt, 16, 32 * ki, 0);
    acc = WMMA_BF16(a, bb, acc);
  }
  #pragma unroll
  for (int v = 0; v < 8; ++v){
    int row = 16 * w + rb + v;
    hid[row * 16 + col] = siluf(acc[v] + b1[row]);
  }
  __syncthreads();
  int t = tid & 15, part = tid >> 4;               // 16 partials x 8 h each
  float s = 0.f;
  for (int h = part * 8; h < part * 8 + 8; ++h) s += hid[h * 16 + t] * w2[h];
  red[part * 16 + t] = s;
  __syncthreads();
  if (tid < 16){
    float o = b2[0];
    for (int p = 0; p < 16; ++p) o += red[p * 16 + tid];
    out[(size_t)b * T_ + t0 + tid] = o;
  }
}

// ---- workspace layout (bytes, all 256-aligned); total ~230.5 MB ------------
#define XBUF_OFF  ((size_t)0)
#define HB_OFF    (XBUF_OFF + (size_t)B_*64*TPK*4)        // bf16 silu(x); aliases xcat
#define FWD_OFF   (HB_OFF   + (size_t)B_*64*TPK*2)
#define INV_OFF   (FWD_OFF  + (size_t)TPK*NMODE2*2)
#define XSPEC_OFF (INV_OFF  + (size_t)NMODE2*TPK*2)
#define YBF_OFF   (XSPEC_OFF+ (size_t)B_*64*NMODE2*4)
#define EMB_OFF   (YBF_OFF  + (size_t)B_*64*NMODE2*2)
#define W1T_OFF   (EMB_OFF  + (size_t)B_*E_*4)
#define W2T_OFF   (W1T_OFF  + (size_t)HL*64*2)
#define SKB_OFF   (W2T_OFF  + (size_t)64*HL*2)
#define P1T_OFF   (SKB_OFF  + (size_t)NL*64*64*2)

extern "C" void kernel_launch(void* const* d_in, const int* in_sizes, int n_in,
                              void* d_out, int out_size, void* d_ws, size_t ws_size,
                              hipStream_t stream){
  (void)in_sizes; (void)n_in; (void)out_size; (void)ws_size;
  const float* vt  = (const float*)d_in[0];
  const float* pp  = (const float*)d_in[1];
  const float* ew1 = (const float*)d_in[2];  const float* eb1 = (const float*)d_in[3];
  const float* ew2 = (const float*)d_in[4];  const float* eb2 = (const float*)d_in[5];
  const float* ew3 = (const float*)d_in[6];  const float* eb3 = (const float*)d_in[7];
  const float* lw1 = (const float*)d_in[8];  const float* lb1 = (const float*)d_in[9];
  const float* lw2 = (const float*)d_in[10]; const float* lb2 = (const float*)d_in[11];
  const float* swr = (const float*)d_in[12]; const float* swi = (const float*)d_in[13];
  const float* skw = (const float*)d_in[14]; const float* skb = (const float*)d_in[15];
  const float* pw1 = (const float*)d_in[16]; const float* pb1 = (const float*)d_in[17];
  const float* pw2 = (const float*)d_in[18]; const float* pb2 = (const float*)d_in[19];

  char* ws = (char*)d_ws;
  float* xbuf  = (float*)(ws + XBUF_OFF);
  bf16*  hb    = (bf16*)(ws + HB_OFF);
  bf16*  xcat  = hb;                     // lift input lives before hb is first written
  bf16*  fwd   = (bf16*)(ws + FWD_OFF);
  bf16*  inv   = (bf16*)(ws + INV_OFF);
  float* xspec = (float*)(ws + XSPEC_OFF);
  bf16*  ybf   = (bf16*)(ws + YBF_OFF);
  float* emb   = (float*)(ws + EMB_OFF);
  bf16*  w1t   = (bf16*)(ws + W1T_OFF);
  bf16*  w2t   = (bf16*)(ws + W2T_OFF);
  bf16*  skbf  = (bf16*)(ws + SKB_OFF);
  bf16*  p1t   = (bf16*)(ws + P1T_OFF);

  k_embed<<<B_, HEMB, 0, stream>>>(pp, ew1, eb1, ew2, eb2, ew3, eb3, emb);
  k_wconv<<<1, 256, 0, stream>>>(lw1, lw2, skw, pw1, w1t, w2t, skbf, p1t);
  k_basis<<<(TPK * NMODE2) / 256, 256, 0, stream>>>(fwd, inv);
  k_xcat<<<dim3(64, B_), 256, 0, stream>>>(vt, emb, xcat);
  k_lift<<<dim3(T_ / 16, B_), 256, 0, stream>>>(xcat, w1t, w2t, lb1, lb2, xbuf);
  k_zeropad<<<(B_ * 64 * (TPK - T_)) / 256, 256, 0, stream>>>(xbuf);

  for (int l = 0; l < NL; ++l){
    k_silu<<<(B_ * 64 * TPK / 4) / 256, 256, 0, stream>>>(xbuf, hb);
    k_dftf<<<B_, 128, 0, stream>>>(hb, fwd, xspec);
    k_specmix<<<64, 256, 0, stream>>>(xspec, swr, swi, ybf, l);
    k_layerout<<<dim3(TPK / 32, B_), 64, 0, stream>>>(hb, skbf, skb, ybf, inv, xbuf, l);
  }
  k_proj<<<dim3(T_ / 16, B_), 256, 0, stream>>>(xbuf, p1t, pb1, pw2, pb2, (float*)d_out);
}
